// HeadAttention_29188597743835
// MI455X (gfx1250) — matrix operand order repaired and checked by reading, not compiled, not run on hardware
//
#include <hip/hip_runtime.h>

// MI455X / gfx1250 flash-attention: QKV proj + causal softmax-attention,
// all matmuls on v_wmma_f32_16x16x32_f16 (wave32).

typedef _Float16 half8  __attribute__((ext_vector_type(8)));
typedef _Float16 half16 __attribute__((ext_vector_type(16)));
typedef float    f32x4  __attribute__((ext_vector_type(4)));
typedef float    f32x8  __attribute__((ext_vector_type(8)));

#define SEQ 4096
#define EMB 1024
#define HID 128
#define NB  8

static __device__ __forceinline__ half16 mk16(half8 lo, half8 hi) {
    half16 r;
#pragma unroll
    for (int i = 0; i < 8; ++i) { r[i] = lo[i]; r[8 + i] = hi[i]; }
    return r;
}

static __device__ __forceinline__ f32x8 vzero8() {
    f32x8 z;
#pragma unroll
    for (int i = 0; i < 8; ++i) z[i] = 0.0f;
    return z;
}

static __device__ __forceinline__ float fast_exp2(float x) {
#if __has_builtin(__builtin_amdgcn_exp2f)
    return __builtin_amdgcn_exp2f(x);
#else
    return exp2f(x);
#endif
}

// ---------------------------------------------------------------------------
// Kernel 1: transpose weights [E,H]f32 -> [H,E]f16; fold log2e/sqrt(H) into Wq
// ---------------------------------------------------------------------------
__global__ void prep_weights(const float* __restrict__ Wq,
                             const float* __restrict__ Wk,
                             const float* __restrict__ Wv,
                             _Float16* __restrict__ WT) {
    int idx = blockIdx.x * blockDim.x + threadIdx.x;   // idx = h*EMB + e
    if (idx >= EMB * HID) return;
    int e = idx & (EMB - 1);
    int h = idx >> 10;
    const float QS = 1.4426950408889634f * 0.08838834764831845f; // log2e / sqrt(128)
    WT[idx]                 = (_Float16)(Wq[e * HID + h] * QS);
    WT[idx + EMB * HID]     = (_Float16)(Wk[e * HID + h]);
    WT[idx + 2 * EMB * HID] = (_Float16)(Wv[e * HID + h]);
}

// ---------------------------------------------------------------------------
// Kernel 2: QKV projection. One wave per 16-row tile of x ([B*S, E] f32).
// Q,K stored row-major [B*S, H] f16; V stored transposed [B, H, S] f16.
// ---------------------------------------------------------------------------
__global__ __launch_bounds__(32) void qkv_proj(const float* __restrict__ x,
                                               const _Float16* __restrict__ WT,
                                               _Float16* __restrict__ Qh,
                                               _Float16* __restrict__ Kh,
                                               _Float16* __restrict__ Vt) {
    const int lane = threadIdx.x;
    const int n    = lane & 15;
    const int hi2  = lane >> 4;
    const int row0 = blockIdx.x * 16;

    f32x8 acc[3][8];
#pragma unroll
    for (int m = 0; m < 3; ++m)
#pragma unroll
        for (int t = 0; t < 8; ++t) acc[m][t] = vzero8();

    for (int c = 0; c < 32; ++c) {           // K-chunks of 32 over E=1024
        // A fragment: rows of x (M = lane&15), f32 -> f16 on the fly
        const float* xr = x + (size_t)(row0 + n) * EMB + c * 32;
        f32x4 a0 = *(const f32x4*)(xr + 8 * hi2);
        f32x4 a1 = *(const f32x4*)(xr + 8 * hi2 + 4);
        f32x4 a2 = *(const f32x4*)(xr + 16 + 8 * hi2);
        f32x4 a3 = *(const f32x4*)(xr + 16 + 8 * hi2 + 4);
        half16 A;
#pragma unroll
        for (int i = 0; i < 4; ++i) {
            A[i]      = (_Float16)a0[i];
            A[4 + i]  = (_Float16)a1[i];
            A[8 + i]  = (_Float16)a2[i];
            A[12 + i] = (_Float16)a3[i];
        }
#pragma unroll
        for (int m = 0; m < 3; ++m) {
#pragma unroll
            for (int ht = 0; ht < 8; ++ht) {
                const _Float16* wp = WT + (size_t)m * EMB * HID
                                   + (size_t)(ht * 16 + n) * EMB + c * 32 + 8 * hi2;
                half16 Bf = mk16(*(const half8*)wp, *(const half8*)(wp + 16));
                acc[m][ht] = __builtin_amdgcn_wmma_f32_16x16x32_f16(
                    false, A, false, Bf, (short)0, acc[m][ht], false, false);
            }
        }
    }

    // Store Q, K row-major f16; V transposed & packed (one b128 per lane/tile)
    const int b    = row0 >> 12;            // row0 / SEQ
    const int sloc = (row0 & (SEQ - 1)) + 8 * hi2;
#pragma unroll
    for (int ht = 0; ht < 8; ++ht) {
#pragma unroll
        for (int r = 0; r < 8; ++r) {
            size_t s = (size_t)row0 + r + 8 * hi2;
            Qh[s * HID + ht * 16 + n] = (_Float16)acc[0][ht][r];
            Kh[s * HID + ht * 16 + n] = (_Float16)acc[1][ht][r];
        }
        half8 vv;
#pragma unroll
        for (int r = 0; r < 8; ++r) vv[r] = (_Float16)acc[2][ht][r];
        *(half8*)(Vt + (size_t)b * HID * SEQ + (size_t)(ht * 16 + n) * SEQ + sloc) = vv;
    }
}

// ---------------------------------------------------------------------------
// Kernel 3: causal flash attention. One wave per (batch, 16-query tile).
// Key tiles of 32 (one WMMA K). Softmax in base-2 (scale folded into Wq).
// ---------------------------------------------------------------------------
__global__ __launch_bounds__(32) void attn(const _Float16* __restrict__ Qh,
                                           const _Float16* __restrict__ Kh,
                                           const _Float16* __restrict__ Vt,
                                           float* __restrict__ out) {
    __shared__ __align__(16) _Float16 plds[16 * 32];   // P tile for D->A transpose

    const int lane  = threadIdx.x;
    const int n     = lane & 15;
    const int hi2   = lane >> 4;
    const int qt    = blockIdx.x;
    const int b     = blockIdx.y;
    const int qbase = qt * 16;
    const size_t rowbase = (size_t)b * SEQ + qbase;

    // Q A-fragments (16 queries x 128 dims = 4 chunks of K=32)
    half16 Aq[4];
#pragma unroll
    for (int c = 0; c < 4; ++c) {
        const _Float16* qp = Qh + (rowbase + n) * HID + c * 32;
        Aq[c] = mk16(*(const half8*)(qp + 8 * hi2),
                     *(const half8*)(qp + 16 + 8 * hi2));
    }

    f32x8 acc[8];
#pragma unroll
    for (int h = 0; h < 8; ++h) acc[h] = vzero8();
    float mrow[8], lrow[8];
#pragma unroll
    for (int r = 0; r < 8; ++r) { mrow[r] = -3.0e38f; lrow[r] = 0.0f; }

    const int jn = (qbase + 47) >> 5;   // key tiles covering keys <= qbase+15
    for (int j = 0; j < jn; ++j) {
        const int k0 = j * 32;

        // ---- scores: S[16q x 32k] via 8 WMMAs ----
        f32x8 s0 = vzero8(), s1 = vzero8();
#pragma unroll
        for (int c = 0; c < 4; ++c) {
            const _Float16* kp0 = Kh + ((size_t)b * SEQ + k0 + n) * HID + c * 32 + 8 * hi2;
            const _Float16* kp1 = Kh + ((size_t)b * SEQ + k0 + 16 + n) * HID + c * 32 + 8 * hi2;
            half16 B0 = mk16(*(const half8*)kp0, *(const half8*)(kp0 + 16));
            half16 B1 = mk16(*(const half8*)kp1, *(const half8*)(kp1 + 16));
            s0 = __builtin_amdgcn_wmma_f32_16x16x32_f16(false, Aq[c], false, B0,
                                                        (short)0, s0, false, false);
            s1 = __builtin_amdgcn_wmma_f32_16x16x32_f16(false, Aq[c], false, B1,
                                                        (short)0, s1, false, false);
        }

        // ---- online softmax (base-2) + causal mask; write P to LDS ----
        float scl[8];
#pragma unroll
        for (int r = 0; r < 8; ++r) {
            const int query = qbase + r + 8 * hi2;
            float v0 = (k0 + n      <= query) ? s0[r] : -3.0e38f;
            float v1 = (k0 + 16 + n <= query) ? s1[r] : -3.0e38f;
            float mx = fmaxf(v0, v1);
            mx = fmaxf(mx, __shfl_xor(mx, 1, 32));
            mx = fmaxf(mx, __shfl_xor(mx, 2, 32));
            mx = fmaxf(mx, __shfl_xor(mx, 4, 32));
            mx = fmaxf(mx, __shfl_xor(mx, 8, 32));
            float mnew = fmaxf(mrow[r], mx);
            float sc = fast_exp2(mrow[r] - mnew);
            float p0 = fast_exp2(v0 - mnew);
            float p1 = fast_exp2(v1 - mnew);
            float rs = p0 + p1;
            rs += __shfl_xor(rs, 1, 32);
            rs += __shfl_xor(rs, 2, 32);
            rs += __shfl_xor(rs, 4, 32);
            rs += __shfl_xor(rs, 8, 32);
            lrow[r] = lrow[r] * sc + rs;
            mrow[r] = mnew;
            scl[r]  = sc;
            plds[(r + 8 * hi2) * 32 + n]      = (_Float16)p0;
            plds[(r + 8 * hi2) * 32 + 16 + n] = (_Float16)p1;
        }
#pragma unroll
        for (int h = 0; h < 8; ++h)
#pragma unroll
            for (int r = 0; r < 8; ++r) acc[h][r] *= scl[r];

        __syncthreads();   // P stores visible
        // P as A-fragment (transpose via LDS): m = lane&15, K split per ISA layout
        half16 Ap = mk16(*(const half8*)&plds[n * 32 + 8 * hi2],
                         *(const half8*)&plds[n * 32 + 16 + 8 * hi2]);
        __syncthreads();   // reads done before next iteration's stores

        // ---- O += P @ V : 8 WMMAs over H-tiles ----
#pragma unroll
        for (int h = 0; h < 8; ++h) {
            const _Float16* vp = Vt + ((size_t)b * HID + h * 16 + n) * SEQ + k0 + 8 * hi2;
            half16 Bv = mk16(*(const half8*)vp, *(const half8*)(vp + 16));
            acc[h] = __builtin_amdgcn_wmma_f32_16x16x32_f16(false, Ap, false, Bv,
                                                            (short)0, acc[h], false, false);
        }
    }

    // ---- epilogue: divide by row sums, store f32 ----
#pragma unroll
    for (int r = 0; r < 8; ++r) {
        float inv = 1.0f / lrow[r];
        size_t orow = (rowbase + r + 8 * hi2) * HID;
#pragma unroll
        for (int h = 0; h < 8; ++h)
            out[orow + h * 16 + n] = acc[h][r] * inv;
    }
}

// ---------------------------------------------------------------------------
extern "C" void kernel_launch(void* const* d_in, const int* in_sizes, int n_in,
                              void* d_out, int out_size, void* d_ws, size_t ws_size,
                              hipStream_t stream) {
    const float* x  = (const float*)d_in[0];
    const float* Wq = (const float*)d_in[1];
    const float* Wk = (const float*)d_in[2];
    const float* Wv = (const float*)d_in[3];
    float* out = (float*)d_out;

    _Float16* Qh = (_Float16*)d_ws;                       // [B*S, H]   f16
    _Float16* Kh = Qh + (size_t)NB * SEQ * HID;           // [B*S, H]   f16
    _Float16* Vt = Kh + (size_t)NB * SEQ * HID;           // [B, H, S]  f16
    _Float16* WT = Vt + (size_t)NB * SEQ * HID;           // 3 x [H, E] f16

    prep_weights<<<(EMB * HID + 255) / 256, 256, 0, stream>>>(Wq, Wk, Wv, WT);
    qkv_proj<<<NB * SEQ / 16, 32, 0, stream>>>(x, WT, Qh, Kh, Vt);
    dim3 g(SEQ / 16, NB);
    attn<<<g, 32, 0, stream>>>(Qh, Kh, Vt, out);
}
